// ESIM_40175124086908
// MI455X (gfx1250) — compile-verified, matrix operand order checked by reading, not run on hardware
//
#include <hip/hip_runtime.h>
#include <hip/hip_bf16.h>

typedef __attribute__((ext_vector_type(16))) _Float16 v16h;
typedef __attribute__((ext_vector_type(8)))  _Float16 v8h;
typedef __attribute__((ext_vector_type(8)))  float    v8f;

#define B_SZ   256
#define T_LEN  128
#define EMB    100
#define KE     128          // EMB padded to 32
#define H_L    300
#define KH     320          // H padded to 32
#define GATES  1200
#define D600   600
#define K6     608          // 600 padded to 32
#define DIN    2400
#define BT     (B_SZ*T_LEN)
#define ROWS   512          // seq A (256) + seq B (256)

// ---------------------------------------------------------------------------
// WMMA fragment helpers (layout per ISA 7.12.2, wave32).
// For both A (MxK) and B (NxK row-major) operands, lane l holds row (l&15);
// each 32-wide K chunk is two contiguous 16B loads at k0+(l>>4)*8 halves and
// k0+16+(l>>4)*8 halves.
// ---------------------------------------------------------------------------
__device__ __forceinline__ v16h load_frag(const _Float16* __restrict__ row,
                                          int k0, int khalf) {
    v8h lo = *(const v8h*)(row + k0 + khalf * 8);
    v8h hi = *(const v8h*)(row + k0 + 16 + khalf * 8);
    return __builtin_shufflevector(lo, hi, 0,1,2,3,4,5,6,7,8,9,10,11,12,13,14,15);
}

__device__ __forceinline__ v8f wmma16(v16h a, v16h b, v8f acc) {
    return __builtin_amdgcn_wmma_f32_16x16x32_f16(false, a, false, b, (short)0, acc,
                                                  false, false);
}

// ---------------------------------------------------------------------------
// Fused LSTM gate GEMM: gates = x_t * Wih^T + h * Whh^T + bias
// Register-blocked 32x48 macro-tile (2x3 WMMA tiles): 6 WMMAs per 32-wide K
// chunk on 5 fragment loads. Fragment row pointers are hoisted out of the K
// loops (loop bodies address with immediate offsets only -> no per-iteration
// 64-bit VALU address math, no WMMA->VALU hazard nops).
// M = 512 = 16 * 32 row-tiles; N = 1200 = 25 * 48 col-tiles (exact, no guards).
// Rows 0..255 = sequence A, 256..511 = sequence B. blockIdx.z = direction.
// K1 is the padded input width (128 for encoder, 2400 for decoder).
// ---------------------------------------------------------------------------
#define TM 2
#define TN 3

__global__ void __launch_bounds__(256)
lstm_gate_gemm(const _Float16* __restrict__ Xa,
               const _Float16* __restrict__ Xb, int K1,
               const _Float16* __restrict__ WihF,
               const _Float16* __restrict__ WihB,
               const _Float16* __restrict__ H,
               const _Float16* __restrict__ WhhF,
               const _Float16* __restrict__ WhhB,
               const float* __restrict__ biasF,
               const float* __restrict__ biasB,
               float* __restrict__ gates, int t) {
    const int dir = blockIdx.z;
    const _Float16* Wih = dir ? WihB : WihF;
    const _Float16* Whh = dir ? WhhB : WhhF;
    const float*   bias = dir ? biasB : biasF;
    const int teff = dir ? (T_LEN - 1 - t) : t;
    const _Float16* Hd = H + (size_t)dir * ROWS * KH;
    float* Gd = gates + (size_t)dir * ROWS * GATES;

    const int wave = threadIdx.x >> 5;
    const int lane = threadIdx.x & 31;
    const int tile = blockIdx.x * 8 + wave;     // 16*25 = 400 macro-tiles per dir
    const int tilesN = GATES / (16 * TN);       // 25
    const int tm = tile / tilesN;               // 0..15 (32-row blocks)
    const int tn = tile % tilesN;               // 0..24 (48-col blocks)
    const int l15 = lane & 15;
    const int khalf = lane >> 4;

    const int r0 = tm * (16 * TM) + l15;        // 32-row tiles never straddle A/B
    const _Float16* xbase = (r0 < B_SZ)
        ? Xa + ((size_t)r0 * T_LEN + teff) * K1
        : Xb + ((size_t)(r0 - B_SZ) * T_LEN + teff) * K1;
    const int c0 = tn * (16 * TN) + l15;

    v8f acc[TM][TN] = {};

    {   // ---- phase 1: x_t * Wih^T (K = K1) --------------------------------
        const _Float16* ar[TM];
        const _Float16* br[TN];
#pragma unroll
        for (int i = 0; i < TM; ++i) ar[i] = xbase + (size_t)i * 16 * T_LEN * K1;
#pragma unroll
        for (int j = 0; j < TN; ++j) br[j] = Wih + (size_t)(c0 + j * 16) * K1;
        for (int k0 = 0; k0 < K1; k0 += 32) {
            v16h a[TM], b[TN];
#pragma unroll
            for (int i = 0; i < TM; ++i) a[i] = load_frag(ar[i], k0, khalf);
#pragma unroll
            for (int j = 0; j < TN; ++j) b[j] = load_frag(br[j], k0, khalf);
#pragma unroll
            for (int i = 0; i < TM; ++i)
#pragma unroll
                for (int j = 0; j < TN; ++j) acc[i][j] = wmma16(a[i], b[j], acc[i][j]);
        }
    }
    {   // ---- phase 2: h * Whh^T (K = KH) ----------------------------------
        const _Float16* ar[TM];
        const _Float16* br[TN];
#pragma unroll
        for (int i = 0; i < TM; ++i) ar[i] = Hd + (size_t)(r0 + i * 16) * KH;
#pragma unroll
        for (int j = 0; j < TN; ++j) br[j] = Whh + (size_t)(c0 + j * 16) * KH;
#pragma unroll
        for (int k0 = 0; k0 < KH; k0 += 32) {
            v16h a[TM], b[TN];
#pragma unroll
            for (int i = 0; i < TM; ++i) a[i] = load_frag(ar[i], k0, khalf);
#pragma unroll
            for (int j = 0; j < TN; ++j) b[j] = load_frag(br[j], k0, khalf);
#pragma unroll
            for (int i = 0; i < TM; ++i)
#pragma unroll
                for (int j = 0; j < TN; ++j) acc[i][j] = wmma16(a[i], b[j], acc[i][j]);
        }
    }

#pragma unroll
    for (int j = 0; j < TN; ++j) {
        int col = tn * (16 * TN) + j * 16 + l15;
        float bc = bias[col];
#pragma unroll
        for (int i = 0; i < TM; ++i) {
            int rbase = tm * (16 * TM) + i * 16 + 8 * khalf;
#pragma unroll
            for (int v = 0; v < 8; ++v)
                Gd[(size_t)(rbase + v) * GATES + col] = acc[i][j][v] + bc;
        }
    }
}

// ---------------------------------------------------------------------------
// LSTM pointwise cell update (torch gate order i,f,g,o).
// ---------------------------------------------------------------------------
__device__ __forceinline__ float sigm(float x) { return 1.0f / (1.0f + __expf(-x)); }

__global__ void lstm_cell_update(const float* __restrict__ gates,
                                 float* __restrict__ c,
                                 _Float16* __restrict__ H,
                                 float* __restrict__ hsA,
                                 float* __restrict__ hsB, int t) {
    int idx = blockIdx.x * blockDim.x + threadIdx.x;
    if (idx >= 2 * ROWS * H_L) return;
    int dir = idx / (ROWS * H_L);
    int r   = (idx / H_L) % ROWS;
    int n   = idx % H_L;
    const float* g = gates + (size_t)dir * ROWS * GATES + (size_t)r * GATES;
    float gi = sigm(g[n]);
    float gf = sigm(g[H_L + n]);
    float gg = tanhf(g[2 * H_L + n]);
    float go = sigm(g[3 * H_L + n]);
    size_t ci = (size_t)dir * ROWS * H_L + (size_t)r * H_L + n;
    float cn = gf * c[ci] + gi * gg;
    c[ci] = cn;
    float h = go * tanhf(cn);
    H[(size_t)dir * ROWS * KH + (size_t)r * KH + n] = (_Float16)h;
    int teff = dir ? (T_LEN - 1 - t) : t;
    float* hs = (r < B_SZ) ? hsA : hsB;
    int b = r & (B_SZ - 1);
    hs[((size_t)b * T_LEN + teff) * D600 + dir * H_L + n] = h;
}

// ---------------------------------------------------------------------------
// Generic batched WMMA GEMM: C[M,N] = A[M,K] * B[N,K]^T (+bias). K % 32 == 0.
// One 16x16 tile per wave; M/N edges handled by clamped loads + guarded stores.
// (Used only for the small attention / fc1 GEMMs.)
// ---------------------------------------------------------------------------
__global__ void wmma_gemm_nt(const _Float16* __restrict__ A,
                             const _Float16* __restrict__ Bm,
                             float* __restrict__ C,
                             const float* __restrict__ bias,
                             int M, int N, int K, int lda, int ldb, int ldc,
                             long sA, long sB, long sC) {
    const int batch = blockIdx.y;
    A  += (size_t)batch * sA;
    Bm += (size_t)batch * sB;
    C  += (size_t)batch * sC;

    const int wave = threadIdx.x >> 5;
    const int lane = threadIdx.x & 31;
    const int tilesN = (N + 15) >> 4;
    const int tilesM = (M + 15) >> 4;
    const int tile = blockIdx.x * 8 + wave;
    if (tile >= tilesM * tilesN) return;
    const int tm = tile / tilesN;
    const int tn = tile % tilesN;
    const int l15 = lane & 15;
    const int khalf = lane >> 4;

    int arow = tm * 16 + l15; if (arow >= M) arow = M - 1;   // clamp (stores guarded)
    int brow = tn * 16 + l15; if (brow >= N) brow = N - 1;
    const _Float16* ar = A  + (size_t)arow * lda;
    const _Float16* br = Bm + (size_t)brow * ldb;

    v8f acc = {};
    for (int k0 = 0; k0 < K; k0 += 32)
        acc = wmma16(load_frag(ar, k0, khalf), load_frag(br, k0, khalf), acc);

    const int col = tn * 16 + l15;
    if (col >= N) return;
    const float bc = bias ? bias[col] : 0.0f;
#pragma unroll
    for (int v = 0; v < 8; ++v) {
        int row = tm * 16 + 8 * khalf + v;
        if (row < M) C[(size_t)row * ldc + col] = acc[v] + bc;
    }
}

// ---------------------------------------------------------------------------
// Small pointwise / staging kernels
// ---------------------------------------------------------------------------
__global__ void embed_gather(const int* __restrict__ tok,
                             const float* __restrict__ embed,
                             _Float16* __restrict__ dst) {
    int idx = blockIdx.x * blockDim.x + threadIdx.x;
    if (idx >= BT * KE) return;
    int bt = idx / KE, k = idx % KE;
    dst[idx] = (k < EMB) ? (_Float16)embed[(size_t)tok[bt] * EMB + k] : (_Float16)0.0f;
}

__global__ void convert_pad(const float* __restrict__ src, _Float16* __restrict__ dst,
                            long R, int K, int Kp) {
    long idx = (long)blockIdx.x * blockDim.x + threadIdx.x;
    if (idx >= R * (long)Kp) return;
    long r = idx / Kp; int k = (int)(idx % Kp);
    dst[idx] = (k < K) ? (_Float16)src[r * K + k] : (_Float16)0.0f;
}

// [B,T,600] f32  ->  [B,600,128] f16 (transposed for the tilde GEMM B-operand)
__global__ void transpose_convert(const float* __restrict__ src, _Float16* __restrict__ dst) {
    long idx = (long)blockIdx.x * blockDim.x + threadIdx.x;
    if (idx >= (long)B_SZ * D600 * T_LEN) return;
    int b = (int)(idx / (D600 * T_LEN));
    int rem = (int)(idx % (D600 * T_LEN));
    int n = rem / T_LEN, t = rem % T_LEN;
    dst[idx] = (_Float16)src[((size_t)b * T_LEN + t) * D600 + n];
}

__global__ void softmax_row(const float* __restrict__ E, _Float16* __restrict__ P) {
    __shared__ float red[T_LEN];
    int row = blockIdx.x, i = threadIdx.x;
    float x = E[(size_t)row * T_LEN + i];
    red[i] = x; __syncthreads();
    for (int s = 64; s > 0; s >>= 1) { if (i < s) red[i] = fmaxf(red[i], red[i + s]); __syncthreads(); }
    float m = red[0]; __syncthreads();
    float e = __expf(x - m);
    red[i] = e; __syncthreads();
    for (int s = 64; s > 0; s >>= 1) { if (i < s) red[i] += red[i + s]; __syncthreads(); }
    P[(size_t)row * T_LEN + i] = (_Float16)(e / red[0]);
}

// softmax over axis 1 (i) of E[b,i,j]; output pre-transposed: P[b,j,i]
__global__ void softmax_col(const float* __restrict__ E, _Float16* __restrict__ P) {
    __shared__ float red[T_LEN];
    int b = blockIdx.x / T_LEN, j = blockIdx.x % T_LEN, i = threadIdx.x;
    float x = E[((size_t)b * T_LEN + i) * T_LEN + j];
    red[i] = x; __syncthreads();
    for (int s = 64; s > 0; s >>= 1) { if (i < s) red[i] = fmaxf(red[i], red[i + s]); __syncthreads(); }
    float m = red[0]; __syncthreads();
    float e = __expf(x - m);
    red[i] = e; __syncthreads();
    for (int s = 64; s > 0; s >>= 1) { if (i < s) red[i] += red[i + s]; __syncthreads(); }
    P[((size_t)b * T_LEN + j) * T_LEN + i] = (_Float16)(e / red[0]);
}

// M = cat(bar, til, bar-til, bar*til) -> f16 [B,T,2400]
__global__ void enhance(const float* __restrict__ bar, const float* __restrict__ til,
                        _Float16* __restrict__ M) {
    long idx = (long)blockIdx.x * blockDim.x + threadIdx.x;
    if (idx >= (long)BT * D600) return;
    long bt = idx / D600; int n = (int)(idx % D600);
    float vb = bar[idx], vt = til[idx];
    _Float16* o = M + bt * DIN;
    o[n]            = (_Float16)vb;
    o[D600 + n]     = (_Float16)vt;
    o[2 * D600 + n] = (_Float16)(vb - vt);
    o[3 * D600 + n] = (_Float16)(vb * vt);
}

__global__ void pool_meanmax(const float* __restrict__ Va, const float* __restrict__ Vb,
                             float* __restrict__ V) {
    int idx = blockIdx.x * blockDim.x + threadIdx.x;
    if (idx >= B_SZ * D600 * 2) return;
    int which = idx / (B_SZ * D600);
    int b = (idx / D600) % B_SZ;
    int n = idx % D600;
    const float* src = which ? Vb : Va;
    float s = 0.0f, mx = -3.4e38f;
    for (int t = 0; t < T_LEN; ++t) {
        float v = src[((size_t)b * T_LEN + t) * D600 + n];
        s += v; mx = fmaxf(mx, v);
    }
    V[(size_t)b * 2400 + which * 1200 + n]       = s * (1.0f / T_LEN);
    V[(size_t)b * 2400 + which * 1200 + 600 + n] = mx;
}

__global__ void fc2_kernel(const float* __restrict__ out1, const float* __restrict__ W,
                           const float* __restrict__ bias, float* __restrict__ out) {
    int idx = blockIdx.x * blockDim.x + threadIdx.x;
    if (idx >= B_SZ * 3) return;
    int b = idx / 3, o = idx % 3;
    float acc = bias[o];
    for (int k = 0; k < 300; ++k) acc += out1[(size_t)b * 300 + k] * W[(size_t)o * 300 + k];
    out[idx] = acc;
}

// ---------------------------------------------------------------------------
// Host launcher
// ---------------------------------------------------------------------------
extern "C" void kernel_launch(void* const* d_in, const int* in_sizes, int n_in,
                              void* d_out, int out_size, void* d_ws, size_t ws_size,
                              hipStream_t stream) {
    // inputs in setup_inputs() order (param dicts flatten Wih, Whh, b)
    const int*   tokA   = (const int*)d_in[0];
    const int*   tokB   = (const int*)d_in[1];
    const float* embed  = (const float*)d_in[2];
    const float* WihEf  = (const float*)d_in[3];
    const float* WhhEf  = (const float*)d_in[4];
    const float* bEf    = (const float*)d_in[5];
    const float* WihEb  = (const float*)d_in[6];
    const float* WhhEb  = (const float*)d_in[7];
    const float* bEb    = (const float*)d_in[8];
    const float* WihDf  = (const float*)d_in[9];
    const float* WhhDf  = (const float*)d_in[10];
    const float* bDf    = (const float*)d_in[11];
    const float* WihDb  = (const float*)d_in[12];
    const float* WhhDb  = (const float*)d_in[13];
    const float* bDb    = (const float*)d_in[14];
    const float* fc1W   = (const float*)d_in[15];
    const float* fc1b   = (const float*)d_in[16];
    const float* fc2W   = (const float*)d_in[17];
    const float* fc2b   = (const float*)d_in[18];
    float* out = (float*)d_out;

    // bump allocator over d_ws (all chunk sizes are multiples of 256 bytes)
    char* base = (char*)d_ws;
    size_t off = 0;
    auto alloc = [&](size_t bytes) { char* p = base + off; off += bytes; return p; };

    _Float16* Xa16    = (_Float16*)alloc((size_t)BT * KE * 2);
    _Float16* Xb16    = (_Float16*)alloc((size_t)BT * KE * 2);
    _Float16* WihEf16 = (_Float16*)alloc((size_t)GATES * KE * 2);
    _Float16* WihEb16 = (_Float16*)alloc((size_t)GATES * KE * 2);
    _Float16* WhhEf16 = (_Float16*)alloc((size_t)GATES * KH * 2);
    _Float16* WhhEb16 = (_Float16*)alloc((size_t)GATES * KH * 2);
    _Float16* WihDf16 = (_Float16*)alloc((size_t)GATES * DIN * 2);
    _Float16* WihDb16 = (_Float16*)alloc((size_t)GATES * DIN * 2);
    _Float16* WhhDf16 = (_Float16*)alloc((size_t)GATES * KH * 2);
    _Float16* WhhDb16 = (_Float16*)alloc((size_t)GATES * KH * 2);
    _Float16* fc1W16  = (_Float16*)alloc((size_t)300 * DIN * 2);
    _Float16* H16     = (_Float16*)alloc((size_t)2 * ROWS * KH * 2);
    float*    Cst     = (float*)   alloc((size_t)2 * ROWS * H_L * 4);
    float*    gates   = (float*)   alloc((size_t)2 * ROWS * GATES * 4);
    float*    Abar    = (float*)   alloc((size_t)BT * D600 * 4);
    float*    Bbar    = (float*)   alloc((size_t)BT * D600 * 4);
    _Float16* Abar16  = (_Float16*)alloc((size_t)BT * K6 * 2);
    _Float16* Bbar16  = (_Float16*)alloc((size_t)BT * K6 * 2);
    _Float16* AbarT16 = (_Float16*)alloc((size_t)B_SZ * D600 * T_LEN * 2);
    _Float16* BbarT16 = (_Float16*)alloc((size_t)B_SZ * D600 * T_LEN * 2);
    float*    E       = (float*)   alloc((size_t)B_SZ * T_LEN * T_LEN * 4);
    _Float16* Pa16    = (_Float16*)alloc((size_t)B_SZ * T_LEN * T_LEN * 2);
    _Float16* Pbt16   = (_Float16*)alloc((size_t)B_SZ * T_LEN * T_LEN * 2);
    float*    Atil    = (float*)   alloc((size_t)BT * D600 * 4);
    float*    Btil    = (float*)   alloc((size_t)BT * D600 * 4);
    _Float16* Ma16    = (_Float16*)alloc((size_t)BT * DIN * 2);
    _Float16* Mb16    = (_Float16*)alloc((size_t)BT * DIN * 2);
    float*    V       = (float*)   alloc((size_t)B_SZ * 2400 * 4);
    _Float16* V16     = (_Float16*)alloc((size_t)B_SZ * 2400 * 2);
    float*    out1    = (float*)   alloc((size_t)B_SZ * 300 * 4);
    float*    Vah     = Abar;   // V_A aliases A_bar (dead after enhance)
    float*    Vbh     = Bbar;   // V_B aliases B_bar

    const int TPB = 256;
    auto blks = [](long n, int tpb) { return (unsigned)((n + tpb - 1) / tpb); };

    // --- stage 0: staging / conversion -------------------------------------
    embed_gather<<<blks((long)BT * KE, TPB), TPB, 0, stream>>>(tokA, embed, Xa16);
    embed_gather<<<blks((long)BT * KE, TPB), TPB, 0, stream>>>(tokB, embed, Xb16);
    convert_pad<<<blks((long)GATES * KE, TPB), TPB, 0, stream>>>(WihEf, WihEf16, GATES, EMB, KE);
    convert_pad<<<blks((long)GATES * KE, TPB), TPB, 0, stream>>>(WihEb, WihEb16, GATES, EMB, KE);
    convert_pad<<<blks((long)GATES * KH, TPB), TPB, 0, stream>>>(WhhEf, WhhEf16, GATES, H_L, KH);
    convert_pad<<<blks((long)GATES * KH, TPB), TPB, 0, stream>>>(WhhEb, WhhEb16, GATES, H_L, KH);
    convert_pad<<<blks((long)GATES * DIN, TPB), TPB, 0, stream>>>(WihDf, WihDf16, GATES, DIN, DIN);
    convert_pad<<<blks((long)GATES * DIN, TPB), TPB, 0, stream>>>(WihDb, WihDb16, GATES, DIN, DIN);
    convert_pad<<<blks((long)GATES * KH, TPB), TPB, 0, stream>>>(WhhDf, WhhDf16, GATES, H_L, KH);
    convert_pad<<<blks((long)GATES * KH, TPB), TPB, 0, stream>>>(WhhDb, WhhDb16, GATES, H_L, KH);
    convert_pad<<<blks((long)300 * DIN, TPB), TPB, 0, stream>>>(fc1W, fc1W16, 300, DIN, DIN);

    // --- stage 1: encoder BiLSTM over both sequences -----------------------
    hipMemsetAsync(H16, 0, (size_t)2 * ROWS * KH * 2, stream);
    hipMemsetAsync(Cst, 0, (size_t)2 * ROWS * H_L * 4, stream);
    for (int t = 0; t < T_LEN; ++t) {
        lstm_gate_gemm<<<dim3(50, 1, 2), TPB, 0, stream>>>(
            Xa16, Xb16, KE, WihEf16, WihEb16, H16, WhhEf16, WhhEb16, bEf, bEb, gates, t);
        lstm_cell_update<<<blks(2L * ROWS * H_L, TPB), TPB, 0, stream>>>(
            gates, Cst, H16, Abar, Bbar, t);
    }

    // --- stage 2: attention ------------------------------------------------
    convert_pad<<<blks((long)BT * K6, TPB), TPB, 0, stream>>>(Abar, Abar16, BT, D600, K6);
    convert_pad<<<blks((long)BT * K6, TPB), TPB, 0, stream>>>(Bbar, Bbar16, BT, D600, K6);
    transpose_convert<<<blks((long)B_SZ * D600 * T_LEN, TPB), TPB, 0, stream>>>(Abar, AbarT16);
    transpose_convert<<<blks((long)B_SZ * D600 * T_LEN, TPB), TPB, 0, stream>>>(Bbar, BbarT16);

    // E[b] = Abar[b] (128x608) * Bbar[b]^T : 64 tiles -> 8 blocks, 256 batches
    wmma_gemm_nt<<<dim3(8, B_SZ), TPB, 0, stream>>>(
        Abar16, Bbar16, E, nullptr, T_LEN, T_LEN, K6, K6, K6, T_LEN,
        (long)T_LEN * K6, (long)T_LEN * K6, (long)T_LEN * T_LEN);

    softmax_row<<<B_SZ * T_LEN, T_LEN, 0, stream>>>(E, Pa16);
    softmax_col<<<B_SZ * T_LEN, T_LEN, 0, stream>>>(E, Pbt16);

    // A_tilde = P_A * B_bar    (M=128, N=600, K=128) ; tiles = 8*38 = 304
    wmma_gemm_nt<<<dim3(38, B_SZ), TPB, 0, stream>>>(
        Pa16, BbarT16, Atil, nullptr, T_LEN, D600, T_LEN, T_LEN, T_LEN, D600,
        (long)T_LEN * T_LEN, (long)D600 * T_LEN, (long)T_LEN * D600);
    // B_tilde = P_B^T * A_bar
    wmma_gemm_nt<<<dim3(38, B_SZ), TPB, 0, stream>>>(
        Pbt16, AbarT16, Btil, nullptr, T_LEN, D600, T_LEN, T_LEN, T_LEN, D600,
        (long)T_LEN * T_LEN, (long)D600 * T_LEN, (long)T_LEN * D600);

    // --- stage 3: enhancement concat ---------------------------------------
    enhance<<<blks((long)BT * D600, TPB), TPB, 0, stream>>>(Abar, Atil, Ma16);
    enhance<<<blks((long)BT * D600, TPB), TPB, 0, stream>>>(Bbar, Btil, Mb16);

    // --- stage 4: decoder BiLSTM (V_A/V_B alias Abar/Bbar) ------------------
    hipMemsetAsync(H16, 0, (size_t)2 * ROWS * KH * 2, stream);
    hipMemsetAsync(Cst, 0, (size_t)2 * ROWS * H_L * 4, stream);
    for (int t = 0; t < T_LEN; ++t) {
        lstm_gate_gemm<<<dim3(50, 1, 2), TPB, 0, stream>>>(
            Ma16, Mb16, DIN, WihDf16, WihDb16, H16, WhhDf16, WhhDb16, bDf, bDb, gates, t);
        lstm_cell_update<<<blks(2L * ROWS * H_L, TPB), TPB, 0, stream>>>(
            gates, Cst, H16, Vah, Vbh, t);
    }

    // --- stage 5: pooling + dense head -------------------------------------
    pool_meanmax<<<blks((long)B_SZ * D600 * 2, TPB), TPB, 0, stream>>>(Vah, Vbh, V);
    convert_pad<<<blks((long)B_SZ * 2400, TPB), TPB, 0, stream>>>(V, V16, B_SZ, 2400, 2400);
    // out1 = V * fc1W^T + fc1b : M=256, N=300, K=2400 -> 16*19 = 304 tiles
    wmma_gemm_nt<<<dim3(38, 1), TPB, 0, stream>>>(
        V16, fc1W16, out1, fc1b, B_SZ, 300, DIN, DIN, DIN, 300, 0L, 0L, 0L);
    fc2_kernel<<<blks((long)B_SZ * 3, TPB), TPB, 0, stream>>>(out1, fc2W, fc2b, out);
}
